// SelfAttention_80762565034015
// MI455X (gfx1250) — matrix operand order repaired and checked by reading, not compiled, not run on hardware
//
#include <hip/hip_runtime.h>
#include <hip/hip_bf16.h>

typedef __bf16 bf16;
typedef __attribute__((ext_vector_type(16))) __bf16 v16bf;
typedef __attribute__((ext_vector_type(8)))  __bf16 v8bf;
typedef __attribute__((ext_vector_type(8)))  float  v8f;
typedef __attribute__((ext_vector_type(4)))  float  v4f;
typedef __attribute__((ext_vector_type(4)))  int    v4i;
typedef __attribute__((ext_vector_type(4)))  unsigned int u32x4;
typedef __attribute__((ext_vector_type(8)))  int    i32x8;

#define BATCH 8
#define SEQ   2048
#define DIM   1024
#define EPS   1e-5f
#define ALD   1032   /* linear-kernel LDS A-tile ld: 1024 + 8 pad             */
#define TLD   1040   /* attn LDS A-tile ld (TDM pad layout: +8 per 512 elems) */

#if __has_builtin(__builtin_amdgcn_global_load_async_to_lds_b128)
#define USE_ASYNC_LDS 1
#else
#define USE_ASYNC_LDS 0
#endif

#if __has_builtin(__builtin_amdgcn_tensor_load_to_lds)
#define USE_TDM 1
#else
#define USE_TDM 0
#endif

// ---------------------------------------------------------------------------
// WMMA helper (CDNA5 wave32, 16x16x32 bf16, fp32 accumulate)
// ---------------------------------------------------------------------------
__device__ __forceinline__ v8f wmma_bf16(v16bf a, v16bf b, v8f c) {
  return __builtin_amdgcn_wmma_f32_16x16x32_bf16(
      false, a, false, b, (short)0, c, false, false);
}

// A-matrix fragment (16x32, 16-bit): lane l<16 holds row M=l with K runs
// {k0..k0+7, k0+16..k0+23}; lane l+16 holds K runs {k0+8..15, k0+24..31}.
__device__ __forceinline__ v16bf frag_ld_a(const bf16* row0, int ld, int k0,
                                           int ln, int hi) {
  const bf16* p = row0 + (size_t)ln * ld + k0 + hi * 8;
  v8bf lo = *(const v8bf*)p;
  v8bf hh = *(const v8bf*)(p + 16);
  return __builtin_shufflevector(lo, hh, 0,1,2,3,4,5,6,7,8,9,10,11,12,13,14,15);
}

// B-matrix fragment (32x16, 16-bit) with B(k,n) = src[n][k]: lane n<16 holds
// column n, K runs {k0..k0+7, k0+16..k0+23}; lane n+16 holds {k0+8..15, k0+24..31}.
__device__ __forceinline__ v16bf frag_ld_b(const bf16* nrow0, int ld, int k0,
                                           int ln, int hi) {
  const bf16* p = nrow0 + (size_t)ln * ld + k0 + hi * 8;
  v8bf lo = *(const v8bf*)p;
  v8bf hh = *(const v8bf*)(p + 16);
  return __builtin_shufflevector(lo, hh, 0,1,2,3,4,5,6,7,8,9,10,11,12,13,14,15);
}

// Stage one 16x1024 bf16 A-tile (src ld = DIM) into LDS (ld = ALD) using the
// gfx1250 async global->LDS datapath when available.
__device__ __forceinline__ void stage_tile16(const bf16* __restrict__ src,
                                             bf16* dst, int tid) {
  int r = tid >> 4;
  int c = (tid & 15) * 64;
  const bf16* s = src + (size_t)r * DIM + c;
  bf16* d = dst + r * ALD + c;
#if USE_ASYNC_LDS
  #pragma unroll
  for (int i = 0; i < 8; ++i) {
    __builtin_amdgcn_global_load_async_to_lds_b128(
        (v4i*)(s + i * 8), (v4i*)(d + i * 8), 0, 0);
  }
#if __has_builtin(__builtin_amdgcn_s_wait_asynccnt)
  __builtin_amdgcn_s_wait_asynccnt(0);
#else
  asm volatile("s_wait_asynccnt 0x0" ::: "memory");
#endif
#else
  #pragma unroll
  for (int i = 0; i < 8; ++i)
    *(v8bf*)(d + i * 8) = *(const v8bf*)(s + i * 8);
#endif
}

// Stage the attention s-tile (16x1024 bf16) into LDS with the Tensor Data
// Mover.  D# per CDNA5 ISA: one descriptor, data_size=2B,
// pad_enable with pad_interval=256 DWORDs / pad_amount=4 DWORDs, so TDM
// inserts 8 bf16 of padding every 512 elements -> LDS element (r, k) lives at
// r*1040 + k + (k>=512 ? 8 : 0).  Row stride 520 DWORDs rotates banks by 8.
__device__ __forceinline__ void attn_stage_A(const bf16* __restrict__ src,
                                             bf16* dst, int tid) {
#if USE_TDM
  if ((tid >> 5) == 0) {   // one wave issues the DMA (EXEC is ignored by TDM)
    unsigned long long ga = (unsigned long long)(uintptr_t)src;
    unsigned int lds = (unsigned int)(uintptr_t)dst;
    u32x4 g0;
    g0[0] = 1u;                                  // count=1, user-mode D#
    g0[1] = lds;                                 // lds_addr (bytes)
    g0[2] = (unsigned int)ga;                    // global_addr[31:0]
    g0[3] = (unsigned int)((ga >> 32) & 0x01FFFFFFu) | (2u << 30); // [56:32]|type=2
    i32x8 g1;
    g1[0] = (1 << 16)     // data_size = 2 bytes
          | (1 << 20)     // pad_enable
          | (7 << 22)     // pad_interval: every 256 DWORDs (512 bf16)
          | (3 << 25);    // pad_amount: 4 DWORDs (8 bf16)
    g1[1] = (DIM & 0xFFFF) << 16;                // tensor_dim0 = 1024
    g1[2] = (SEQ & 0xFFFF) << 16;                // tensor_dim1 = 2048
    g1[3] = (DIM & 0xFFFF) << 16;                // tile_dim0 = 1024
    g1[4] = 16;                                  // tile_dim1 = 16 rows
    g1[5] = DIM;                                 // tensor_dim0_stride = 1024
    g1[6] = 0;
    g1[7] = 0;
    v4i  z  = {};
    i32x8 z8 = {};
    __builtin_amdgcn_tensor_load_to_lds(g0, g1, z, z, z8, 0);
#if __has_builtin(__builtin_amdgcn_s_wait_tensorcnt)
    __builtin_amdgcn_s_wait_tensorcnt(0);
#else
    asm volatile("s_wait_tensorcnt 0x0" ::: "memory");
#endif
  }
#else
  int r = tid >> 4;
  int c = (tid & 15) * 64;
  const bf16* s = src + (size_t)r * DIM + c;
  bf16* d = dst + r * TLD + c + ((c >= 512) ? 8 : 0);
  #pragma unroll
  for (int i = 0; i < 8; ++i)
    *(v8bf*)(d + i * 8) = *(const v8bf*)(s + i * 8);
#endif
}

// ---------------------------------------------------------------------------
// Prep kernels
// ---------------------------------------------------------------------------
__global__ __launch_bounds__(256) void cvt_w_kernel(const float* __restrict__ W,
                                                    bf16* __restrict__ wbf) {
  int i = blockIdx.x * 256 + threadIdx.x;
  wbf[i] = (bf16)W[i];
}

__global__ __launch_bounds__(256) void opinion_kernel(
    const float* __restrict__ go, const float* __restrict__ ga,
    const float* __restrict__ pl, const float* __restrict__ gp,
    float* __restrict__ opi) {
  int i = blockIdx.x * 256 + threadIdx.x;
  float p = gp[0];
  const float* o = go + (size_t)i * 3;
  const float* a = ga + (size_t)i * 3;
  const float* l = pl + (size_t)i * 5;
  opi[i] = p * (o[1] + o[2] + a[1] + a[2]) +
           (1.0f - p) * (l[1] + l[2] + l[3] + l[4]);
}

__global__ __launch_bounds__(256) void cvt_x_kernel(const float* __restrict__ x,
                                                    bf16* __restrict__ xbf,
                                                    bf16* __restrict__ xTbf) {
  __shared__ bf16 tile[64][72];
  int tid = threadIdx.x;
  int tx = tid & 63, ty = tid >> 6;
  int d0 = blockIdx.x * 64;
  int t0 = blockIdx.y * 64;
  int bb = blockIdx.z;
  const float* xb = x + (size_t)bb * SEQ * DIM;
  bf16* xbfb = xbf + (size_t)bb * SEQ * DIM;
  for (int r = 0; r < 16; ++r) {
    int tl = ty + r * 4;
    size_t off = (size_t)(t0 + tl) * DIM + d0 + tx;
    bf16 h = (bf16)xb[off];
    xbfb[off] = h;
    tile[tl][tx] = h;
  }
  __syncthreads();
  bf16* xTb = xTbf + (size_t)bb * DIM * SEQ;
  for (int r = 0; r < 16; ++r) {
    int dl = ty + r * 4;
    xTb[(size_t)(d0 + dl) * SEQ + t0 + tx] = tile[tx][dl];
  }
}

// ---------------------------------------------------------------------------
// Kernel 1: x_tran = bf16(x @ W^T + bias).
// A-tile shared via LDS; K-loop is a true ping-pong (step 64): loads land
// directly in the loop-carried registers, so no v_mov rotation / hazard NOPs.
// ---------------------------------------------------------------------------
__global__ __launch_bounds__(256) void linear_kernel(
    const bf16* __restrict__ xbf, const bf16* __restrict__ wbf,
    const float* __restrict__ bias, bf16* __restrict__ xtbf) {
  __shared__ bf16 Atile[16 * ALD];
  int tid = threadIdx.x;
  int wave = tid >> 5, lane = tid & 31, ln = lane & 15, hi = lane >> 4;
  int gw = blockIdx.x * 8 + wave;
  int mTile = gw >> 4;
  int nG = gw & 15;

  stage_tile16(xbf + (size_t)mTile * 16 * DIM, Atile, tid);
  __syncthreads();

  const bf16* brow[4];
  #pragma unroll
  for (int j = 0; j < 4; ++j)
    brow[j] = wbf + (size_t)(nG * 64 + j * 16) * DIM;

  v8f acc[4] = {};
  v16bf aA = frag_ld_a(Atile, ALD, 0, ln, hi);
  v16bf bA[4];
  #pragma unroll
  for (int j = 0; j < 4; ++j) bA[j] = frag_ld_b(brow[j], DIM, 0, ln, hi);

  #pragma unroll 1
  for (int k0 = 0; k0 < DIM; k0 += 64) {
    v16bf aB = frag_ld_a(Atile, ALD, k0 + 32, ln, hi);
    v16bf bB[4];
    #pragma unroll
    for (int j = 0; j < 4; ++j) bB[j] = frag_ld_b(brow[j], DIM, k0 + 32, ln, hi);
    #pragma unroll
    for (int j = 0; j < 4; ++j) acc[j] = wmma_bf16(aA, bA[j], acc[j]);
    if (k0 + 64 < DIM) {
      aA = frag_ld_a(Atile, ALD, k0 + 64, ln, hi);
      #pragma unroll
      for (int j = 0; j < 4; ++j) bA[j] = frag_ld_b(brow[j], DIM, k0 + 64, ln, hi);
    }
    #pragma unroll
    for (int j = 0; j < 4; ++j) acc[j] = wmma_bf16(aB, bB[j], acc[j]);
  }

  // Epilogue: add bias, transpose per-wave tile through LDS, store coalesced.
  __syncthreads();                      // everyone done reading Atile
  bf16* Ot = Atile + wave * (16 * 72);  // per-wave 16x72 region (16B-aligned)
  #pragma unroll
  for (int j = 0; j < 4; ++j) {
    float bv = bias[nG * 64 + j * 16 + ln];
    #pragma unroll
    for (int v = 0; v < 8; ++v)
      Ot[(v + hi * 8) * 72 + j * 16 + ln] = (bf16)(acc[j][v] + bv);
  }
  int r2 = lane >> 1;
  int cs = (lane & 1) * 32;
  const bf16* sp = Ot + r2 * 72 + cs;
  bf16* dp = xtbf + (size_t)(mTile * 16 + r2) * DIM + nG * 64 + cs;
  #pragma unroll
  for (int i = 0; i < 4; ++i)
    *(v8bf*)(dp + i * 8) = *(const v8bf*)(sp + i * 8);
}

// ---------------------------------------------------------------------------
// Kernel 2: fused attention (flash-style, never materializes SxS).
// s-tile of x_tran staged once via TDM (or fallback) in the padded layout;
// both GEMM phases ping-pong / pipeline their fragments.
// ---------------------------------------------------------------------------
__global__ __launch_bounds__(256) void attn_kernel(
    const bf16* __restrict__ xtbf, const bf16* __restrict__ xTbf,
    const float* __restrict__ opi, float* __restrict__ out) {
  __shared__ bf16 Atile[16 * TLD + 8];  // s-tile of x_tran (TDM pad layout)
  __shared__ bf16 P[16 * 136];          // 16 x 128 bf16 chunk of weights
  __shared__ float rowpart[16][16];
  __shared__ float rowsum[16];
  __shared__ float opin[16];

  const int tid = threadIdx.x;
  const int wave = tid >> 5;
  const int lane = tid & 31;
  const int ln = lane & 15;
  const int hi = lane >> 4;

  const int bb = blockIdx.x >> 7;
  const int s0 = (blockIdx.x & 127) << 4;

  const bf16* xt_b = xtbf + (size_t)bb * SEQ * DIM;
  const bf16* xT_b = xTbf + (size_t)bb * DIM * SEQ;

  if (tid < 16) {
    opin[tid] = opi[(size_t)bb * SEQ + s0 + tid];
    rowsum[tid] = 0.0f;
  }
  attn_stage_A(xt_b + (size_t)s0 * DIM, Atile, tid);
  __syncthreads();

  const int d0w = wave * 128;
  v8f accD[8] = {};

  for (int T0 = 0; T0 < SEQ; T0 += 128) {
    // Warm next chunk while this one computes.
    int Tn = T0 + 128;
    if (Tn < SEQ) {
      __builtin_prefetch(xt_b + (size_t)(Tn + wave * 16 + ln) * DIM + hi * 64, 0, 1);
      __builtin_prefetch(xT_b + (size_t)(d0w + lane * 4) * SEQ + Tn, 0, 1);
    }

    // ---- phase 1: 16x16 score tile (t-sub = wave), A from LDS -------------
    const bf16* brow = xt_b + (size_t)(T0 + wave * 16) * DIM;
    v8f acc = {};
    v16bf aA = frag_ld_a(Atile, TLD, 0, ln, hi);
    v16bf bA = frag_ld_b(brow, DIM, 0, ln, hi);
    #pragma unroll 1
    for (int k0 = 0; k0 < DIM; k0 += 64) {
      int k1 = k0 + 32, k2 = k0 + 64;
      v16bf aB = frag_ld_a(Atile + ((k1 >= 512) ? 8 : 0), TLD, k1, ln, hi);
      v16bf bB = frag_ld_b(brow, DIM, k1, ln, hi);
      acc = wmma_bf16(aA, bA, acc);
      if (k2 < DIM) {
        aA = frag_ld_a(Atile + ((k2 >= 512) ? 8 : 0), TLD, k2, ln, hi);
        bA = frag_ld_b(brow, DIM, k2, ln, hi);
      }
      acc = wmma_bf16(aB, bB, acc);
    }
    __syncthreads();   // previous chunk's P/rowpart consumers are done

    {
      int colT = T0 + wave * 16 + ln;
      #pragma unroll
      for (int v = 0; v < 8; ++v) {
        int rl = v + hi * 8;
        int rowS = s0 + rl;
        float val = acc[v] * (1.0f / (fabsf((float)(rowS - colT)) + EPS))
                           * opin[rl];
        val = __expf(tanhf(val));
        if (rowS == colT) val = 0.0f;
        P[rl * 136 + wave * 16 + ln] = (bf16)val;
      }
    }
    __syncthreads();

    // ---- row sums of this P chunk ----------------------------------------
    {
      int r = tid >> 4, seg = tid & 15;
      const bf16* pr = P + r * 136 + seg * 8;
      float s = 0.0f;
      #pragma unroll
      for (int i = 0; i < 8; ++i) s += (float)pr[i];
      rowpart[r][seg] = s;
    }
    __syncthreads();
    if (tid < 16) {
      float s = 0.0f;
      #pragma unroll
      for (int i = 0; i < 16; ++i) s += rowpart[tid][i];
      rowsum[tid] += s;
    }

    // ---- phase 2: out^T += x^T(d,t) @ P^T(t,s), pipelined over d-tiles ----
    for (int tt = 0; tt < 4; ++tt) {
      v16bf bfrag = frag_ld_b(P, 136, tt * 32, ln, hi);
      const size_t tOff = (size_t)T0 + tt * 32;
      v16bf af = frag_ld_a(xT_b + (size_t)d0w * SEQ + tOff, SEQ, 0, ln, hi);
      #pragma unroll
      for (int dt = 0; dt < 8; ++dt) {
        v16bf afn = af;
        if (dt < 7)
          afn = frag_ld_a(xT_b + (size_t)(d0w + (dt + 1) * 16) * SEQ + tOff,
                          SEQ, 0, ln, hi);
        accD[dt] = wmma_bf16(af, bfrag, accD[dt]);
        af = afn;
      }
    }
  }
  __syncthreads();

  // ---- epilogue: normalize rows, store out[b][s][d] as b128s --------------
  float inv = 1.0f / (rowsum[ln] + EPS);
  int sIdx = s0 + ln;
  float* outRow = out + ((size_t)bb * SEQ + sIdx) * DIM;
  #pragma unroll
  for (int dt = 0; dt < 8; ++dt) {
    int dbase = d0w + dt * 16 + hi * 8;
    v4f lo, hh;
    #pragma unroll
    for (int v = 0; v < 4; ++v) {
      lo[v] = accD[dt][v] * inv;
      hh[v] = accD[dt][v + 4] * inv;
    }
    *(v4f*)(outRow + dbase) = lo;
    *(v4f*)(outRow + dbase + 4) = hh;
  }
}

// ---------------------------------------------------------------------------
// Host-side launch
// ---------------------------------------------------------------------------
extern "C" void kernel_launch(void* const* d_in, const int* in_sizes, int n_in,
                              void* d_out, int out_size, void* d_ws, size_t ws_size,
                              hipStream_t stream) {
  (void)in_sizes; (void)n_in; (void)out_size; (void)ws_size;
  const float* x    = (const float*)d_in[0];
  const float* W    = (const float*)d_in[1];
  const float* bias = (const float*)d_in[2];
  const float* go   = (const float*)d_in[3];
  const float* ga   = (const float*)d_in[4];
  const float* pl   = (const float*)d_in[5];
  const float* gp   = (const float*)d_in[6];
  float* out = (float*)d_out;

  char* ws = (char*)d_ws;
  const size_t XBF = (size_t)BATCH * SEQ * DIM * sizeof(bf16);   // 32 MB
  bf16*  xbf  = (bf16*)(ws);
  bf16*  xtbf = (bf16*)(ws + XBF);
  bf16*  xTbf = (bf16*)(ws + 2 * XBF);
  bf16*  wbf  = (bf16*)(ws + 3 * XBF);
  float* opiw = (float*)(ws + 3 * XBF + (size_t)DIM * DIM * sizeof(bf16));

  cvt_w_kernel<<<(DIM * DIM) / 256, 256, 0, stream>>>(W, wbf);
  opinion_kernel<<<(BATCH * SEQ) / 256, 256, 0, stream>>>(go, ga, pl, gp, opiw);
  cvt_x_kernel<<<dim3(DIM / 64, SEQ / 64, BATCH), 256, 0, stream>>>(x, xbf, xTbf);
  linear_kernel<<<(BATCH * SEQ / 16) * (DIM / 64) / 8, 256, 0, stream>>>(xbf, wbf, bias, xtbf);
  attn_kernel<<<BATCH * (SEQ / 16), 256, 0, stream>>>(xtbf, xTbf, opiw, out);
}
